// BinaryLinear_18683107737780
// MI455X (gfx1250) — matrix operand order repaired and checked by reading, not compiled, not run on hardware
//
#include <hip/hip_runtime.h>

typedef __attribute__((ext_vector_type(16))) _Float16 v16h;
typedef __attribute__((ext_vector_type(8)))  _Float16 v8h;
typedef __attribute__((ext_vector_type(8)))  float    v8f;

#define K_DIM 4096
#define N_DIM 4096
#define LDS_STRIDE 40   // 32 halves + 8 halves pad (80B row pitch -> conflict-free b128 reads)

union H4 { _Float16 h[4]; unsigned long long u; };

// Packed f16 sign: two f32 -> two f16 in one u32.
// sign(w)*1.0h == 0x3C00 | signbit. (Exact-zero weight maps to +/-1 instead of 0:
// measure-zero event for f32 normal data, saves ~10x VALU vs compare/select chain.)
static __device__ __forceinline__ unsigned int sgnpk2(float a, float b) {
    unsigned int ua = __builtin_bit_cast(unsigned int, a);
    unsigned int ub = __builtin_bit_cast(unsigned int, b);
    return 0x3C003C00u | ((ua >> 16) & 0x8000u) | (ub & 0x80000000u);
}

// ---------------- Pass 1: per-block partial sums of |W| (deterministic tree) --------------
__global__ __launch_bounds__(256) void binlin_absum_partial(const float* __restrict__ w,
                                                            float* __restrict__ partial) {
    __shared__ float s[256];
    const int t = threadIdx.x;
    const float4* __restrict__ w4 = (const float4*)(w + (size_t)blockIdx.x * 8192);
    float acc = 0.f;
#pragma unroll
    for (int i = 0; i < 8; ++i) {
        float4 v = w4[t + 256 * i];
        acc += fabsf(v.x) + fabsf(v.y) + fabsf(v.z) + fabsf(v.w);
    }
    s[t] = acc;
    __syncthreads();
    for (int o = 128; o > 0; o >>= 1) {
        if (t < o) s[t] += s[t + o];
        __syncthreads();
    }
    if (t == 0) partial[blockIdx.x] = s[0];
}

// ---------------- Pass 2: final reduce -> scale = mean(|W|) --------------
__global__ __launch_bounds__(256) void binlin_absum_final(const float* __restrict__ partial,
                                                          float* __restrict__ scale) {
    __shared__ float s[256];
    const int t = threadIdx.x;
    float acc = 0.f;
    for (int i = t; i < 2048; i += 256) acc += partial[i];
    s[t] = acc;
    __syncthreads();
    for (int o = 128; o > 0; o >>= 1) {
        if (t < o) s[t] += s[t + o];
        __syncthreads();
    }
    if (t == 0) scale[0] = s[0] * (1.0f / 16777216.0f);   // / (4096*4096)
}

// ---------------- Pass 3: WMMA GEMM: out = scale * (x @ sign(W)^T) --------------
// Block: 256 threads = 8 waves, macro-tile 128(M) x 128(N), K-step 32.
// Wave grid 2x4: each wave computes 64x32 = 4x2 tiles of 16x16 f32 accumulators.
// Double-buffered LDS pipeline: next slab's global loads overlap current slab's WMMAs.
__global__ __launch_bounds__(256) void binlin_gemm(const float* __restrict__ x,
                                                   const float* __restrict__ w,
                                                   const float* __restrict__ scaleptr,
                                                   float* __restrict__ out) {
    __shared__ __align__(16) _Float16 Alds[2][128 * LDS_STRIDE];   // [m][k] f16, 2x10KB
    __shared__ __align__(16) _Float16 Blds[2][128 * LDS_STRIDE];   // [n][k] f16 (=sign(W)), 2x10KB

    const int tid   = threadIdx.x;
    const int lane  = tid & 31;
    const int wave  = tid >> 5;
    const int waveM = wave >> 2;       // 0..1
    const int waveN = wave & 3;        // 0..3
    const int h     = lane >> 4;       // wave half
    const int r     = lane & 15;

    const int tileM = blockIdx.y * 128;
    const int tileN = blockIdx.x * 128;

    const int loadRow = tid >> 3;      // 0..31
    const int loadC4  = tid & 7;       // 0..7  (float4 column within 32-wide k slab)

    const float4* __restrict__ x4 = (const float4*)x;
    const float4* __restrict__ w4 = (const float4*)w;

    v8f acc[4][2];
#pragma unroll
    for (int ms = 0; ms < 4; ++ms)
#pragma unroll
        for (int ns = 0; ns < 2; ++ns)
#pragma unroll
            for (int q = 0; q < 8; ++q) acc[ms][ns][q] = 0.0f;

    // Per-thread global row bases (float4 units) for the 4 rows this thread stages.
    size_t abase[4], bbase[4];
#pragma unroll
    for (int i = 0; i < 4; ++i) {
        abase[i] = (size_t)(tileM + loadRow + 32 * i) * (K_DIM / 4) + loadC4;
        bbase[i] = (size_t)(tileN + loadRow + 32 * i) * (K_DIM / 4) + loadC4;
    }

    float4 areg[4], breg[4];

    // convert + store staged registers into LDS buffer b
    auto stash = [&](int b) {
#pragma unroll
        for (int i = 0; i < 4; ++i) {
            const int row = loadRow + 32 * i;
            // A: f32 -> f16 (v_cvt_pk_f16_f32 pairs)
            H4 ah;
            ah.h[0] = (_Float16)areg[i].x; ah.h[1] = (_Float16)areg[i].y;
            ah.h[2] = (_Float16)areg[i].z; ah.h[3] = (_Float16)areg[i].w;
            *(unsigned long long*)&Alds[b][row * LDS_STRIDE + 4 * loadC4] = ah.u;
            // B: packed bit-trick sign -> +/-1.0h
            const unsigned long long bu =
                (unsigned long long)sgnpk2(breg[i].x, breg[i].y) |
                ((unsigned long long)sgnpk2(breg[i].z, breg[i].w) << 32);
            *(unsigned long long*)&Blds[b][row * LDS_STRIDE + 4 * loadC4] = bu;
        }
    };

    // fragment assembly (per ISA VGPR layout) + 8 WMMAs on LDS buffer b
    auto compute = [&](int b) {
        v16h af[4];
        v16h bf[2];
#pragma unroll
        for (int ms = 0; ms < 4; ++ms) {
            // A 16x32 f16: halves 0..7 = K[8h .. 8h+7], halves 8..15 = K[16+8h .. 16+8h+7]
            const int arow = waveM * 64 + ms * 16 + r;
            const _Float16* p = &Alds[b][arow * LDS_STRIDE + 8 * h];
            v8h lo = *(const v8h*)p;          // 16B aligned (row pitch 80B, offset 16h)
            v8h hi = *(const v8h*)(p + 16);
#pragma unroll
            for (int j = 0; j < 8; ++j) { af[ms][j] = lo[j]; af[ms][j + 8] = hi[j]; }
        }
#pragma unroll
        for (int ns = 0; ns < 2; ++ns) {
            // B 32x16 f16: halves 0..7 = K[8h .. 8h+7], halves 8..15 = K[16+8h .. 16+8h+7]
            const int brow = waveN * 32 + ns * 16 + r;
            const _Float16* p = &Blds[b][brow * LDS_STRIDE + 8 * h];
            v8h lo = *(const v8h*)p;
            v8h hi = *(const v8h*)(p + 16);
#pragma unroll
            for (int j = 0; j < 8; ++j) { bf[ns][j] = lo[j]; bf[ns][j + 8] = hi[j]; }
        }
#pragma unroll
        for (int ms = 0; ms < 4; ++ms)
#pragma unroll
            for (int ns = 0; ns < 2; ++ns)
                acc[ms][ns] = __builtin_amdgcn_wmma_f32_16x16x32_f16(
                    /*neg_a=*/false, af[ms], /*neg_b=*/false, bf[ns],
                    /*c_mod=*/(short)0, acc[ms][ns],
                    /*reuse_a=*/false, /*reuse_b=*/false);
    };

    // ---- prologue: stage slab k=0 into buffer 0
#pragma unroll
    for (int i = 0; i < 4; ++i) { areg[i] = x4[abase[i]]; breg[i] = w4[bbase[i]]; }
    stash(0);
    __syncthreads();

    // ---- steady state: load slab kk+32 into regs, compute slab kk, store regs -> other buffer
    int buf = 0;
    for (int kk = 0; kk < K_DIM - 32; kk += 32) {
        const int nb4 = (kk + 32) >> 2;   // next slab, float4 units
#pragma unroll
        for (int i = 0; i < 4; ++i) {
            areg[i] = x4[abase[i] + nb4];
            breg[i] = w4[bbase[i] + nb4];
        }
        // speculative HBM->L2 warm-up two slabs ahead (gfx1250 global_prefetch_b8)
        __builtin_prefetch(&x4[abase[0] + nb4 + 8], 0, 0);
        __builtin_prefetch(&w4[bbase[0] + nb4 + 8], 0, 0);

        compute(buf);          // WMMAs overlap the in-flight global loads
        stash(buf ^ 1);        // waits on loads, fills the other buffer
        __syncthreads();
        buf ^= 1;
    }
    compute(buf);              // peeled last slab

    // ---- epilogue: scale by mean(|W|), store f32
    const float scale = *scaleptr;
#pragma unroll
    for (int ms = 0; ms < 4; ++ms) {
#pragma unroll
        for (int ns = 0; ns < 2; ++ns) {
            const int col = tileN + waveN * 32 + ns * 16 + r;
#pragma unroll
            for (int q = 0; q < 8; ++q) {
                const int row = tileM + waveM * 64 + ms * 16 + h * 8 + q;   // C/D: VGPR q -> M = q + 8h
                out[(size_t)row * N_DIM + col] = acc[ms][ns][q] * scale;
            }
        }
    }
}

// ---------------- host-side launch --------------
extern "C" void kernel_launch(void* const* d_in, const int* in_sizes, int n_in,
                              void* d_out, int out_size, void* d_ws, size_t ws_size,
                              hipStream_t stream) {
    const float* x = (const float*)d_in[0];       // [8,2048,4096] f32 -> M=16384 rows
    const float* w = (const float*)d_in[1];       // [4096,4096]   f32
    float* out = (float*)d_out;
    float* ws  = (float*)d_ws;                    // [0..2047] partials, [2048] scale

    const int M = in_sizes[0] / K_DIM;            // 16384

    binlin_absum_partial<<<2048, 256, 0, stream>>>(w, ws);
    binlin_absum_final<<<1, 256, 0, stream>>>(ws, ws + 2048);

    dim3 grid(N_DIM / 128, M / 128);              // 32 x 128 blocks
    binlin_gemm<<<grid, 256, 0, stream>>>(x, w, ws + 2048, out);
}
